// Radon3D_84670985273674
// MI455X (gfx1250) — matrix operand order repaired and checked by reading, not compiled, not run on hardware
//
#include <hip/hip_runtime.h>
#include <hip/hip_fp16.h>

#define G  320
#define NA 180
#define DT 5    // d-tiles per wave: B fragment built once, used by DT WMMAs

typedef __attribute__((ext_vector_type(16))) _Float16 v16h;
typedef __attribute__((ext_vector_type(8)))  _Float16 v8h;
typedef __attribute__((ext_vector_type(4)))  _Float16 v4h;
typedef __attribute__((ext_vector_type(8)))  float    v8f;

// Pre-pass: convert the f32 volume to f16 once (A operands are f16 anyway).
__global__ __launch_bounds__(256)
void cvt_f32_to_f16_kernel(const float* __restrict__ x, _Float16* __restrict__ xh)
{
    const size_t i = ((size_t)blockIdx.x * 256 + (size_t)threadIdx.x) * 4;
    const float4 v = *(const float4*)(x + i);
    v4h o = { (_Float16)v.x, (_Float16)v.y, (_Float16)v.z, (_Float16)v.w };
    *(v4h*)(xh + i) = o;
}

// One wave (32 lanes) per workgroup.
// blockIdx.x = angle, blockIdx.y = group of DT d-tiles, blockIdx.z = w-tile.
// Per input row y: build B(32x16) bilinear-weight fragment once (f32 LDS,
// native ds_add_f32 scatter), then acc_dt += A_dt(16x32) @ B for DT d-tiles.
template <typename TA>
__global__ __launch_bounds__(32)
void radon3d_wmma_kernel(const TA* __restrict__ x, float* __restrict__ out)
{
    const int a    = blockIdx.x;
    const int d0   = blockIdx.y * (16 * DT);
    const int w0   = blockIdx.z << 4;
    const int lane = threadIdx.x;
    const int n    = lane & 15;   // output column within tile / A row m
    const int hi   = lane >> 4;   // half-wave select

    __shared__ float Bbuf[16][32];   // [column][K-slot] weight accumulation

    // Reference converts deg->rad twice.
    const float ang = (float)a * (3.14159265f / 180.0f) * (3.141592f / 180.0f);
    const float c  = cosf(ang);            // [0.9985, 1]
    const float s  = sinf(ang);            // [0, 0.0545]
    const float gf = (float)G;
    // ix = c*w + s*h + K1 ; iy = c*h - s*w + K2   (align_corners=False)
    const float K1 = 0.5f * (c + s + gf - c * gf - s * gf - 1.0f);
    const float K2 = 0.5f * (c - s + gf + s * gf - c * gf - 1.0f);

    const float wnf  = (float)(w0 + n);
    const float t    = K2 - s * wnf;                 // iy(h) = c*h + t
    const float tmid = K2 - s * ((float)w0 + 8.0f);  // wave-uniform center
    const float invc = 1.0f / c;

    v8f acc[DT];
    #pragma unroll
    for (int dt = 0; dt < DT; ++dt) acc[dt] = (v8f){};

    // 32-bit element offsets (volume < 2^31 elements) -> saddr+voffset loads
    unsigned rowoff[DT];
    #pragma unroll
    for (int dt = 0; dt < DT; ++dt)
        rowoff[dt] = (unsigned)(d0 + 16 * dt + n) * (unsigned)(G * G);

    for (int y = 0; y < G; ++y) {
        const float yf = (float)y;

        // Wave-uniform x-window base, 8-element aligned, clamped so the
        // 32-wide window is always fully inside the row. Margin analysis
        // guarantees every in-volume tap still lands at k in [0,31];
        // out-of-volume taps are weight-zeroed below.
        const float hc = (yf - tmid) * invc;
        int xb = (int)floorf(c * (float)w0 + s * hc + K1) - 2;
        xb &= ~7;
        xb = xb < 0 ? 0 : (xb > G - 32 ? G - 32 : xb);

        // ---- zero B buffer (512 floats, 16 per lane) ----
        {
            float4* bz = (float4*)(&Bbuf[0][0]);
            const float4 z4 = make_float4(0.f, 0.f, 0.f, 0.f);
            bz[lane * 4 + 0] = z4;
            bz[lane * 4 + 1] = z4;
            bz[lane * 4 + 2] = z4;
            bz[lane * 4 + 3] = z4;
        }
        // Single wave: DS ops are in program order -> no barrier needed.

        // ---- branchless weight scatter (native ds_add_f32) ----
        // h candidates with floor(iy) in {y-1,y} are hb..hb+2.
        // lanes 0-15 take {hb, hb+1}; lanes 16-31 take {hb+2, hb+3} (hb+3 is
        // provably weightless). Invalid taps get weight 0 and a clamped slot;
        // ds_add of 0.0 is harmless.
        const int hb = (int)ceilf((yf - 1.0f - t) * invc);
        #pragma unroll
        for (int ci = 0; ci < 2; ++ci) {
            const int h     = hb + (hi ? 2 : 0) + ci;
            const float hf  = (float)h;
            const float iy  = c * hf + t;
            const float iyf = floorf(iy);
            const int   iy0 = (int)iyf;
            const float fy  = iy - iyf;
            float wyv = (iy0 == y) ? (1.0f - fy)
                      : ((iy0 == y - 1) ? fy : 0.0f);
            wyv = ((unsigned)h < (unsigned)G) ? wyv : 0.0f;

            const float ix  = c * wnf + s * hf + K1;
            const float ixf = floorf(ix);
            const int   ix0 = (int)ixf;
            const float fx  = ix - ixf;
            const int   k0  = ix0 - xb;
            const int   k1  = k0 + 1;

            float w0v = wyv * (1.0f - fx);
            float w1v = wyv * fx;
            w0v = ((unsigned)ix0       < (unsigned)G && (unsigned)k0 < 32u) ? w0v : 0.0f;
            w1v = ((unsigned)(ix0 + 1) < (unsigned)G && (unsigned)k1 < 32u) ? w1v : 0.0f;
            const int k0c = k0 < 0 ? 0 : (k0 > 31 ? 31 : k0);
            const int k1c = k1 < 0 ? 0 : (k1 > 31 ? 31 : k1);

            atomicAdd(&Bbuf[n][k0c], w0v);   // ds_add_f32
            atomicAdd(&Bbuf[n][k1c], w1v);
        }

        // ---- B fragment: lane (n,hi) holds K = hi*8 + e (+8 if e>=8), column n ----
        v16h bfrag;
        {
            const float* bp = &Bbuf[n][hi * 8];
            #pragma unroll
            for (int e = 0; e < 16; ++e)
                bfrag[e] = (_Float16)bp[e < 8 ? e : e + 8];           // ds_load_b128 + cvt_pk
        }

        // ---- per d-tile: A fragment + WMMA (B reused DT times) ----
        #pragma unroll
        for (int dt = 0; dt < DT; ++dt) {
            v16h afrag;
            if constexpr (sizeof(TA) == 2) {
                // f16 volume: element e -> k = e + 8*hi (+8 if e>=8)
                const _Float16* rp = (const _Float16*)x + rowoff[dt] + xb;  // 16B aligned
                v8h a0 = *(const v8h*)(rp + 8 * hi);
                v8h a1 = *(const v8h*)(rp + 16 + 8 * hi);
                afrag = __builtin_shufflevector(a0, a1,
                            0, 1, 2, 3, 4, 5, 6, 7, 8, 9, 10, 11, 12, 13, 14, 15);
                if (y + 1 < G) __builtin_prefetch(rp + G, 0, 1);
            } else {
                const float4* rp = (const float4*)((const float*)x + rowoff[dt] + xb);
                const float4 q0 = rp[2 * hi + 0];
                const float4 q1 = rp[2 * hi + 1];
                const float4 q2 = rp[4 + 2 * hi + 0];
                const float4 q3 = rp[4 + 2 * hi + 1];
                afrag[0]  = (_Float16)q0.x; afrag[1]  = (_Float16)q0.y;
                afrag[2]  = (_Float16)q0.z; afrag[3]  = (_Float16)q0.w;
                afrag[4]  = (_Float16)q1.x; afrag[5]  = (_Float16)q1.y;
                afrag[6]  = (_Float16)q1.z; afrag[7]  = (_Float16)q1.w;
                afrag[8]  = (_Float16)q2.x; afrag[9]  = (_Float16)q2.y;
                afrag[10] = (_Float16)q2.z; afrag[11] = (_Float16)q2.w;
                afrag[12] = (_Float16)q3.x; afrag[13] = (_Float16)q3.y;
                afrag[14] = (_Float16)q3.z; afrag[15] = (_Float16)q3.w;
            }
            acc[dt] = __builtin_amdgcn_wmma_f32_16x16x32_f16(
                false, afrag, false, bfrag, (short)0, acc[dt], false, false);
            rowoff[dt] += G;
        }
    }

    // ---- store: VGPR v -> row v + 8*hi of each tile, column n; layout (D,W,A) ----
    #pragma unroll
    for (int dt = 0; dt < DT; ++dt) {
        #pragma unroll
        for (int v = 0; v < 8; ++v) {
            const int m = d0 + 16 * dt + v + 8 * hi;
            out[((size_t)m * (size_t)G + (size_t)(w0 + n)) * (size_t)NA + (size_t)a]
                = acc[dt][v];
        }
    }
}

extern "C" void kernel_launch(void* const* d_in, const int* in_sizes, int n_in,
                              void* d_out, int out_size, void* d_ws, size_t ws_size,
                              hipStream_t stream)
{
    (void)in_sizes; (void)n_in; (void)out_size;
    const float* x = (const float*)d_in[0];
    float* out     = (float*)d_out;

    const size_t nvox = (size_t)G * (size_t)G * (size_t)G;
    dim3 grid(NA, G / (16 * DT), G / 16);   // 180 x 4 x 20 single-wave WGs
    dim3 block(32);

    if (ws_size >= nvox * sizeof(_Float16)) {
        _Float16* xh = (_Float16*)d_ws;
        cvt_f32_to_f16_kernel<<<dim3((unsigned)(nvox / 4 / 256)), dim3(256), 0, stream>>>(x, xh);
        radon3d_wmma_kernel<_Float16><<<grid, block, 0, stream>>>(xh, out);
    } else {
        radon3d_wmma_kernel<float><<<grid, block, 0, stream>>>(x, out);
    }
}
